// SubsetsSampleWeightedFormula_71347996721714
// MI455X (gfx1250) — matrix operand order repaired and checked by reading, not compiled, not run on hardware
//
#include <hip/hip_runtime.h>

// ---------------- problem constants ----------------
#define BB   32
#define AA   48
#define GFD  256
#define SS   4096
#define PP   8
#define SBN  512
#define NEL  5
#define OHSZ 20
#define FEW  100          // formula encoding width
#define IDD  512
#define K1D  384          // GF+FE=356 padded to multiple of 32
#define NROWS (BB * SS)   // 131072

typedef __bf16 bf16_t;
typedef bf16_t v16bf __attribute__((ext_vector_type(16)));
typedef bf16_t v8bf  __attribute__((ext_vector_type(8)));
typedef float  v8f   __attribute__((ext_vector_type(8)));
typedef float  v4f   __attribute__((ext_vector_type(4)));

// -------- CDNA5 async global->LDS copy (ASYNCcnt path), guarded fallback --------
#if defined(__gfx1250__) && __has_builtin(__builtin_amdgcn_global_load_async_to_lds_b128)
#define HAVE_ASYNC_LDS 1
#else
#define HAVE_ASYNC_LDS 0
#endif

// builtin signature (leaked by clang diagnostic): param0 = GCC-vector int4 in AS1,
// param1 = int4 in AS3, then two integer immediates (offset, cpol).
typedef int v4i_gcc __attribute__((vector_size(16)));
typedef __attribute__((address_space(1))) v4i_gcc* gv4p;
typedef __attribute__((address_space(3))) v4i_gcc* lv4p;

__device__ __forceinline__ void cp_async16(void* lds, const void* g) {
#if HAVE_ASYNC_LDS
  __builtin_amdgcn_global_load_async_to_lds_b128(
      (gv4p)(unsigned long long)(g),
      (lv4p)(unsigned int)(unsigned long long)(lds), 0, 0);
#else
  *(v4f*)lds = *(const v4f*)g;
#endif
}

__device__ __forceinline__ void wait_async_all() {
#if HAVE_ASYNC_LDS
#if __has_builtin(__builtin_amdgcn_s_wait_asynccnt)
  __builtin_amdgcn_s_wait_asynccnt(0);
#else
  asm volatile("s_wait_asynccnt 0x0" ::: "memory");
#endif
#endif
}

// pin the software-pipelined DS-read / WMMA interleave (no-op if unsupported)
__device__ __forceinline__ void sched_pipeline_16x16(void) {
#if __has_builtin(__builtin_amdgcn_sched_group_barrier)
  __builtin_amdgcn_sched_group_barrier(0x100, 8, 0);  // DS read x8 (grp0)
  __builtin_amdgcn_sched_group_barrier(0x100, 8, 0);  // DS read x8 (grp1)
  __builtin_amdgcn_sched_group_barrier(0x008, 4, 0);  // WMMA x4  (grp0)
  __builtin_amdgcn_sched_group_barrier(0x100, 8, 0);  // DS read x8 (grp2)
  __builtin_amdgcn_sched_group_barrier(0x008, 4, 0);  // WMMA x4  (grp1)
  __builtin_amdgcn_sched_group_barrier(0x100, 8, 0);  // DS read x8 (grp3)
  __builtin_amdgcn_sched_group_barrier(0x008, 4, 0);  // WMMA x4  (grp2)
  __builtin_amdgcn_sched_group_barrier(0x008, 4, 0);  // WMMA x4  (grp3)
#endif
}

// ---------------- weight prep: fp32 [K][N] -> bf16 [N][K] (transposed, padded) ----------------
__global__ __launch_bounds__(256) void prep_weights(
    const float* __restrict__ W1, const float* __restrict__ W2a, const float* __restrict__ W2b,
    bf16_t* __restrict__ W1t, bf16_t* __restrict__ W2at, bf16_t* __restrict__ W2bt)
{
  int id = blockIdx.x * 256 + threadIdx.x;
  const int n1 = 512 * K1D;      // 196608
  const int n2 = 512 * 512;      // 262144
  if (id < n1) {
    int n = id / K1D, k = id % K1D;
    float v = (k < (GFD + FEW)) ? W1[k * 512 + n] : 0.f;
    W1t[id] = (bf16_t)v;
  } else if (id < n1 + n2) {
    int j = id - n1; int n = j / 512, k = j % 512;
    W2at[j] = (bf16_t)W2a[k * 512 + n];
  } else if (id < n1 + 2 * n2) {
    int j = id - n1 - n2; int n = j / 512, k = j % 512;
    W2bt[j] = (bf16_t)W2b[k * 512 + n];
  }
}

// ---------------- stage A: subset-weighted mean + LN + thermometer -> x0 bf16 [NROWS][K1D] ----------------
__global__ __launch_bounds__(256) void stage_a(
    const float* __restrict__ feat,    // [B,A,GF]
    const float* __restrict__ vmask,   // [B,A]
    const float* __restrict__ eloh,    // [B,A,NE]
    const int*   __restrict__ subs,    // [B,S,A] 0/1
    const float* __restrict__ g1, const float* __restrict__ b1v,
    bf16_t* __restrict__ X0)
{
  __shared__ float featm[AA * GFD];       // masked features, 48KB
  __shared__ float ml[AA];
  __shared__ float ohl[AA * NEL];
  __shared__ float sbuf[16 * AA];
  __shared__ float xbuf[16 * GFD];        // 16KB
  __shared__ float psum[256], psq[256];
  __shared__ float sizeL[16];
  __shared__ int   cntL[16 * NEL];

  int t  = threadIdx.x;
  int b  = blockIdx.x >> 8;               // 256 row-tiles per batch
  int s0 = (blockIdx.x & 255) * 16;

  if (t < AA) ml[t] = vmask[b * AA + t];
  for (int i = t; i < AA * GFD; i += 256) {
    int a = i >> 8;
    featm[i] = feat[b * AA * GFD + i] * vmask[b * AA + a];
  }
  for (int i = t; i < AA * NEL; i += 256) ohl[i] = eloh[b * AA * NEL + i];
  for (int i = t; i < 16 * AA; i += 256) {
    int r = i / AA, a = i % AA;
    sbuf[i] = (float)subs[((long)b * SS + s0 + r) * AA + a];
  }
  __syncthreads();

  int r = t >> 4, c = t & 15;             // thread owns row r, 16 cols at c*16
  float acc[16];
  #pragma unroll
  for (int i = 0; i < 16; i++) acc[i] = 0.f;
  for (int a = 0; a < AA; a++) {
    float s = sbuf[r * AA + a];
    const float* fr = &featm[a * GFD + c * 16];
    #pragma unroll
    for (int i = 0; i < 16; i++) acc[i] = fmaf(s, fr[i], acc[i]);
  }
  if (c == 0) {
    float size = 1e-4f;
    float cnt[NEL] = {0.f, 0.f, 0.f, 0.f, 0.f};
    for (int a = 0; a < AA; a++) {
      float s = sbuf[r * AA + a];
      size += s * ml[a];
      #pragma unroll
      for (int e = 0; e < NEL; e++) cnt[e] += s * ohl[a * NEL + e];
    }
    sizeL[r] = size;
    #pragma unroll
    for (int e = 0; e < NEL; e++) {
      int ci = (int)cnt[e];
      cntL[r * NEL + e] = min(max(ci, 0), OHSZ - 1);
    }
  }
  #pragma unroll
  for (int i = 0; i < 16; i++) xbuf[r * GFD + c * 16 + i] = acc[i];
  __syncthreads();

  float inv = 1.f / sizeL[r];
  float ps = 0.f, pq = 0.f;
  #pragma unroll
  for (int i = 0; i < 16; i++) {
    float v = xbuf[r * GFD + c * 16 + i] * inv;
    ps += v; pq += v * v;
  }
  psum[t] = ps; psq[t] = pq;
  __syncthreads();
  float sum = 0.f, sq = 0.f;
  #pragma unroll
  for (int i = 0; i < 16; i++) { sum += psum[r * 16 + i]; sq += psq[r * 16 + i]; }
  float mean = sum * (1.f / GFD);
  float var  = sq * (1.f / GFD) - mean * mean;
  float rstd = rsqrtf(var + 1e-5f);

  long row = (long)b * SS + s0 + r;
  bf16_t* outp = X0 + row * K1D;
  #pragma unroll
  for (int i = 0; i < 16; i++) {
    int col = c * 16 + i;
    float v = xbuf[r * GFD + col] * inv;
    v = (v - mean) * rstd * g1[col] + b1v[col];
    outp[col] = (bf16_t)v;
  }
  #pragma unroll
  for (int i = 0; i < 8; i++) {           // cols 256..383: formula + zero pad
    int f = c * 8 + i;
    float v = 0.f;
    if (f < FEW) {
      int e = f / OHSZ, th = f % OHSZ;
      v = (th >= cntL[r * NEL + e]) ? 1.f : 0.f;
    }
    outp[GFD + f] = (bf16_t)v;
  }
}

// ---------------- bf16 WMMA GEMM + bias + relu ----------------
// Y[M,512] = relu(X[M,Kdim] * Wt^T + bias), Wt stored [N=512][Kdim].
// Block = 64 rows x 512 cols, 8 wave32. Weights double-buffered in LDS via
// async-to-LDS; A fragments prefetched from global one K-step ahead.
// K-loop manually unrolled x2 so all buffer indices are compile-time.
#define XS 40   // LDS row stride in halfs (80B: 16B aligned, bank-spread)
__global__ __launch_bounds__(256) void gemm_bias_relu(
    const bf16_t* __restrict__ X, int Kdim,
    const bf16_t* __restrict__ Wt,
    const float*  __restrict__ bias,
    bf16_t* __restrict__ Y)
{
  __shared__ bf16_t wlds[2][512 * XS];   // 2 x 40KB ping-pong

  int t = threadIdx.x;
  int lane = t & 31, wv = t >> 5;
  int mwave = (wv & 3) * 16;             // 4 M-tiles across waves
  int nhalf = wv >> 2;                   // 2 N-halves of 256 cols
  int khalf = lane >> 4, l16 = lane & 15;
  long rowBase = (long)blockIdx.x * 64;
  const bf16_t* xrow = X + (rowBase + mwave + l16) * Kdim;

  v8f acc[16];
  const v8f vzero = {0.f, 0.f, 0.f, 0.f, 0.f, 0.f, 0.f, 0.f};
  #pragma unroll
  for (int i = 0; i < 16; i++) acc[i] = vzero;

  // stage one 512x32 W slice into LDS buffer (8 x 16B chunks per thread)
  auto issueW = [&](bf16_t* buf, int k0) {
    #pragma unroll
    for (int i = 0; i < 8; i++) {
      int cI = t + i * 256;
      int n = cI >> 2, ch = cI & 3;
      cp_async16(&buf[n * XS + ch * 8],
                 Wt + (long)n * Kdim + k0 + ch * 8);
    }
  };
  // A fragment direct from global: VGPR0-3 K=khalf*8+0..7, VGPR4-7 K=16+khalf*8+0..7
  auto loadA = [&](int k0) -> v16bf {
    v8bf lo = *(const v8bf*)(xrow + k0 + khalf * 8);
    v8bf hi = *(const v8bf*)(xrow + k0 + 16 + khalf * 8);
    return __builtin_shufflevector(lo, hi,
        0, 1, 2, 3, 4, 5, 6, 7, 8, 9, 10, 11, 12, 13, 14, 15);
  };
  // B fragment: col = l16 of tile; lane reads K = khalf*8+0..7, 16+khalf*8+0..7
  auto loadB = [&](const bf16_t* wb, int nt) -> v16bf {
    int n = nhalf * 256 + nt * 16 + l16;
    v8bf lo = *(const v8bf*)&wb[n * XS + khalf * 8];
    v8bf hi = *(const v8bf*)&wb[n * XS + 16 + khalf * 8];
    return __builtin_shufflevector(lo, hi,
        0, 1, 2, 3, 4, 5, 6, 7, 8, 9, 10, 11, 12, 13, 14, 15);
  };
  // 16 WMMAs on one K-slice; B fragments in explicit groups-of-4 so 8
  // ds_load_b128 for group g+1 are in flight behind the 4 WMMAs of group g.
  auto computeStep = [&](const bf16_t* wb, v16bf a) {
    v16bf bf0[4], bf1[4];
    #pragma unroll
    for (int i = 0; i < 4; i++) bf0[i] = loadB(wb, i);
    #pragma unroll
    for (int g = 0; g < 4; g++) {
      v16bf* curB = (g & 1) ? bf1 : bf0;
      v16bf* nxtB = (g & 1) ? bf0 : bf1;
      if (g < 3) {
        #pragma unroll
        for (int i = 0; i < 4; i++) nxtB[i] = loadB(wb, (g + 1) * 4 + i);
      }
      #pragma unroll
      for (int i = 0; i < 4; i++)
        acc[g * 4 + i] = __builtin_amdgcn_wmma_f32_16x16x32_bf16(
            false, a, false, curB[i], (short)0, acc[g * 4 + i], false, false);
    }
    sched_pipeline_16x16();   // enforce DS-read / WMMA interleave
  };

  int nk = Kdim >> 5;                    // 12 or 16: always even
  v16bf aEven, aOdd;
  issueW(wlds[0], 0);
  aEven = loadA(0);

  for (int k = 0; k < nk; k += 2) {
    wait_async_all();          // own async writes for slice k complete
    __syncthreads();           // all waves' writes visible; prev compute done
    if (k + 1 < nk) {
      issueW(wlds[1], (k + 1) * 32);
      aOdd = loadA((k + 1) * 32);
    }
    computeStep(wlds[0], aEven);

    wait_async_all();          // slice k+1 complete
    __syncthreads();
    if (k + 2 < nk) {
      issueW(wlds[0], (k + 2) * 32);
      aEven = loadA((k + 2) * 32);
    }
    computeStep(wlds[1], aOdd);
  }

  // epilogue: D layout lane<16 -> M=j, lane>=16 -> M=8+j; N = l16 within tile
  #pragma unroll
  for (int nt = 0; nt < 16; nt++) {
    int n = nhalf * 256 + nt * 16 + l16;
    float bv = bias[n];
    #pragma unroll
    for (int j = 0; j < 8; j++) {
      long row = rowBase + mwave + khalf * 8 + j;
      float v = acc[nt][j] + bv;
      Y[row * IDD + n] = (bf16_t)fmaxf(v, 0.f);
    }
  }
}

// ---------------- layernorm(512) + score dot ----------------
__global__ __launch_bounds__(256) void ln_score(
    const bf16_t* __restrict__ X3,
    const float* __restrict__ g2, const float* __restrict__ b2,
    const float* __restrict__ Ws, const float* __restrict__ bs,
    float* __restrict__ scores)
{
  int t = threadIdx.x, lane = t & 31, wv = t >> 5;
  long row = (long)blockIdx.x * 8 + wv;
  v16bf xv = *(const v16bf*)(X3 + row * IDD + lane * 16);
  float v[16];
  float s = 0.f, sq = 0.f;
  #pragma unroll
  for (int i = 0; i < 16; i++) { v[i] = (float)xv[i]; s += v[i]; sq += v[i] * v[i]; }
  #pragma unroll
  for (int o = 16; o >= 1; o >>= 1) { s += __shfl_xor(s, o, 32); sq += __shfl_xor(sq, o, 32); }
  float mean = s * (1.f / IDD);
  float var  = sq * (1.f / IDD) - mean * mean;
  float rstd = rsqrtf(var + 1e-5f);
  float dot = 0.f;
  #pragma unroll
  for (int i = 0; i < 16; i++) {
    int e = lane * 16 + i;
    dot += ((v[i] - mean) * rstd * g2[e] + b2[e]) * Ws[e];
  }
  #pragma unroll
  for (int o = 16; o >= 1; o >>= 1) dot += __shfl_xor(dot, o, 32);
  if (lane == 0) scores[row] = dot + bs[0];
}

// ---------------- softmax over S per batch; writes probs region of d_out ----------------
__global__ __launch_bounds__(256) void softmax_k(
    const float* __restrict__ scores, float* __restrict__ out)
{
  __shared__ float red[256];
  int b = blockIdx.x, t = threadIdx.x;
  const float* sc = scores + (long)b * SS;
  float m = -1e30f;
  for (int i = t; i < SS; i += 256) m = fmaxf(m, sc[i]);
  red[t] = m; __syncthreads();
  for (int o = 128; o >= 1; o >>= 1) { if (t < o) red[t] = fmaxf(red[t], red[t + o]); __syncthreads(); }
  m = red[0]; __syncthreads();
  float sum = 0.f;
  for (int i = t; i < SS; i += 256) sum += __expf(sc[i] - m);
  red[t] = sum; __syncthreads();
  for (int o = 128; o >= 1; o >>= 1) { if (t < o) red[t] += red[t + o]; __syncthreads(); }
  float inv = 1.f / red[0];
  float* probs = out + BB * SBN + (long)b * SS;
  for (int i = t; i < SS; i += 256) probs[i] = __expf(sc[i] - m) * inv;
}

// ---------------- weighted histogram into spect region of d_out ----------------
__global__ __launch_bounds__(256) void hist_k(
    const float* __restrict__ peaks, float* __restrict__ out)
{
  __shared__ float h[SBN];
  int b = blockIdx.x, t = threadIdx.x;
  h[t] = 0.f; h[t + 256] = 0.f;
  __syncthreads();
  const float* probs = out + BB * SBN + (long)b * SS;
  for (int s = t; s < SS; s += 256) {
    float p = probs[s];
    const float* pk = peaks + ((long)b * SS + s) * PP * 2;
    #pragma unroll
    for (int i = 0; i < PP; i++) {
      int bin = (int)rintf(pk[i * 2]);
      bin = min(max(bin, 0), SBN - 1);
      atomicAdd(&h[bin], pk[i * 2 + 1] * p);   // ds_add_f32
    }
  }
  __syncthreads();
  out[(long)b * SBN + t] = h[t];
  out[(long)b * SBN + 256 + t] = h[t + 256];
}

// ---------------- host launcher ----------------
extern "C" void kernel_launch(void* const* d_in, const int* in_sizes, int n_in,
                              void* d_out, int out_size, void* d_ws, size_t ws_size,
                              hipStream_t stream) {
  (void)in_sizes; (void)n_in; (void)out_size; (void)ws_size;
  const float* feat  = (const float*)d_in[0];
  const float* vmask = (const float*)d_in[1];
  const float* eloh  = (const float*)d_in[2];
  /* d_in[3] adj_oh unused */
  const int*   subs  = (const int*)d_in[4];
  const float* peaks = (const float*)d_in[5];
  const float* ln1g  = (const float*)d_in[6];
  const float* ln1b  = (const float*)d_in[7];
  const float* W1    = (const float*)d_in[8];
  const float* b1    = (const float*)d_in[9];
  const float* W2a   = (const float*)d_in[10];
  const float* b2a   = (const float*)d_in[11];
  const float* W2b   = (const float*)d_in[12];
  const float* b2b   = (const float*)d_in[13];
  const float* ln2g  = (const float*)d_in[14];
  const float* ln2b  = (const float*)d_in[15];
  const float* Ws    = (const float*)d_in[16];
  const float* bs    = (const float*)d_in[17];
  float* out = (float*)d_out;

  char* ws = (char*)d_ws;
  bf16_t* W1t   = (bf16_t*)(ws);                 //  384 KB
  bf16_t* W2at  = (bf16_t*)(ws + 393216);        //  512 KB
  bf16_t* W2bt  = (bf16_t*)(ws + 917504);        //  512 KB
  float*  scores= (float*) (ws + 1441792);       //  512 KB
  char* bufA = ws + 2097152;                     //  128 MB
  char* bufB = bufA + 134217728;                 //  128 MB
  bf16_t* x0 = (bf16_t*)bufA;                    // [131072,384]
  bf16_t* x1 = (bf16_t*)bufB;                    // [131072,512]
  bf16_t* x2 = (bf16_t*)bufA;                    // ping-pong
  bf16_t* x3 = (bf16_t*)bufB;

  prep_weights<<<2816, 256, 0, stream>>>(W1, W2a, W2b, W1t, W2at, W2bt);
  stage_a<<<BB * (SS / 16), 256, 0, stream>>>(feat, vmask, eloh, subs, ln1g, ln1b, x0);
  gemm_bias_relu<<<NROWS / 64, 256, 0, stream>>>(x0, K1D, W1t, b1, x1);
  gemm_bias_relu<<<NROWS / 64, 256, 0, stream>>>(x1, IDD, W2at, b2a, x2);
  gemm_bias_relu<<<NROWS / 64, 256, 0, stream>>>(x2, IDD, W2bt, b2b, x3);
  ln_score<<<NROWS / 8, 256, 0, stream>>>(x3, ln2g, ln2b, Ws, bs, scores);
  softmax_k<<<BB, 256, 0, stream>>>(scores, out);
  hist_k<<<BB, 256, 0, stream>>>(peaks, out);
}
